// DeltaNet_50448685859253
// MI455X (gfx1250) — compile-verified, matrix operand order checked by reading, not run on hardware
//
#include <hip/hip_runtime.h>
#include <hip/hip_bf16.h>
#include <stdint.h>

// ---------------- problem constants ----------------
#define B_    4
#define L_    2048
#define D_    1024
#define H_    4
#define DK_   128
#define DV_   256
#define CS_   16
#define KD_   512     // H*DK
#define VD_   1024    // H*DV
#define CONV_ 4
#define NC_   (L_/CS_)   // 128 chunks
#define DVS_  32         // DV slice per delta block
#define NSL_  (DV_/DVS_) // 8 slices

typedef __attribute__((ext_vector_type(16))) _Float16 v16h;
typedef __attribute__((ext_vector_type(8)))  float    v8f;
typedef __attribute__((ext_vector_type(2)))  float    v2f;
typedef __attribute__((ext_vector_type(4)))  unsigned int u32x4;
typedef __attribute__((ext_vector_type(4)))  int      i32x4;

// ---------------- CDNA5 async global->LDS support (guarded) ----------------
#if defined(__has_builtin)
#  if __has_builtin(__builtin_amdgcn_global_load_async_to_lds_b128)
#    define HAVE_ASYNC_LDS 1
#  endif
#endif

__device__ __forceinline__ void cp16_lds(void* lds_dst, const void* gsrc) {
#ifdef HAVE_ASYNC_LDS
  __builtin_amdgcn_global_load_async_to_lds_b128(
      (__attribute__((address_space(1))) i32x4*)gsrc,
      (__attribute__((address_space(3))) i32x4*)lds_dst, 0, 0);
#else
  *(u32x4*)lds_dst = *(const u32x4*)gsrc;
#endif
}

#ifdef HAVE_ASYNC_LDS
#  if __has_builtin(__builtin_amdgcn_s_wait_asynccnt)
#    define WAIT_ASYNC(n) __builtin_amdgcn_s_wait_asynccnt(n)
#  else
#    define WAIT_ASYNC(n) asm volatile("s_wait_asynccnt %0" :: "i"(n) : "memory")
#  endif
#else
#  define WAIT_ASYNC(n) ((void)0)
#endif

__device__ __forceinline__ v8f wmma_f16(v16h a, v16h b, v8f c) {
  return __builtin_amdgcn_wmma_f32_16x16x32_f16(false, a, false, b, (short)0, c, false, false);
}
__device__ __forceinline__ v8f wmma_f32(v2f a, v2f b, v8f c) {
  return __builtin_amdgcn_wmma_f32_16x16x4_f32(false, a, false, b, (short)0, c, false, false);
}
__device__ __forceinline__ float silu_f(float x) { return x / (1.f + __expf(-x)); }

// ---------------- conv + SiLU, also emit f16 copy ----------------
__global__ __launch_bounds__(256) void conv_silu_kernel(
    const float* __restrict__ hs, const float* __restrict__ cw,
    float* __restrict__ x, _Float16* __restrict__ xh)
{
  int idx = blockIdx.x * 256 + threadIdx.x;
  if (idx >= B_ * L_ * D_) return;
  int d = idx % D_;
  int l = (idx / D_) % L_;
  int b = idx / (D_ * L_);
  float acc = 0.f;
#pragma unroll
  for (int j = 0; j < CONV_; ++j) {
    int t = l + j - (CONV_ - 1);
    if (t >= 0) acc += hs[((size_t)b * L_ + t) * D_ + d] * cw[d * CONV_ + j];
  }
  float y = silu_f(acc);
  x[idx] = y;
  xh[idx] = (_Float16)y;
}

// ---------------- f32 -> f16 cast ----------------
__global__ __launch_bounds__(256) void cvt_f32_f16_kernel(
    const float* __restrict__ in, _Float16* __restrict__ out, int n)
{
  int i = blockIdx.x * 256 + threadIdx.x;
  if (i < n) out[i] = (_Float16)in[i];
}

// ---------------- beta = sigmoid(x @ Wb^T), one wave per row ----------------
__global__ __launch_bounds__(256) void beta_kernel(
    const float* __restrict__ x, const float* __restrict__ Wb, float* __restrict__ beta)
{
  int row = blockIdx.x * 8 + (threadIdx.x >> 5);
  int lane = threadIdx.x & 31;
  if (row >= B_ * L_) return;
  const float* xr = x + (size_t)row * D_;
  for (int h = 0; h < H_; ++h) {
    const float* w = Wb + (size_t)h * D_;
    float s = 0.f;
    for (int i = lane; i < D_; i += 32) s += xr[i] * w[i];
    for (int off = 16; off; off >>= 1) s += __shfl_down(s, off, 32);
    s = __shfl(s, 0, 32);
    if (lane == 0) beta[(size_t)row * H_ + h] = 1.f / (1.f + __expf(-s));
  }
}

// ---------------- tiled f16 WMMA GEMM: C[M,N] = A[M,K] @ W[N,K]^T ----------------
// double-buffered LDS, async global->LDS staging overlapped with WMMA
__global__ __launch_bounds__(256) void gemm_f16_kernel(
    const _Float16* __restrict__ A, const _Float16* __restrict__ W,
    float* __restrict__ C, int M, int N, int K)
{
  __shared__ _Float16 lA[2][128][40];
  __shared__ _Float16 lB[2][128][40];
  int tid = threadIdx.x;
  int wave = tid >> 5, lane = tid & 31;
  int bm = blockIdx.y * 128, bn = blockIdx.x * 128;

  v8f acc[8];
#pragma unroll
  for (int i = 0; i < 8; ++i) acc[i] = (v8f){0.f,0.f,0.f,0.f,0.f,0.f,0.f,0.f};

  // stage one K-tile (32 halves wide) into buffer `buf`: 4 async B128 ops/thread
  auto stage = [&](int buf, int k0) {
    for (int i = tid; i < 128 * 4; i += 256) {
      int r = i >> 2, cch = (i & 3) * 8;
      cp16_lds(&lA[buf][r][cch], A + (size_t)(bm + r) * K + k0 + cch);
    }
    for (int i = tid; i < 128 * 4; i += 256) {
      int r = i >> 2, cch = (i & 3) * 8;
      cp16_lds(&lB[buf][r][cch], W + (size_t)(bn + r) * K + k0 + cch);
    }
  };

  stage(0, 0);

  for (int k0 = 0; k0 < K; k0 += 32) {
    int cur = (k0 >> 5) & 1;
    bool more = (k0 + 32) < K;
    if (more) stage(cur ^ 1, k0 + 32);
    if (more) { WAIT_ASYNC(8); } else { WAIT_ASYNC(0); }
    __syncthreads();

    int m = lane & 15;
    int klo = (lane < 16) ? 0 : 8;
    v16h a;
#pragma unroll
    for (int v = 0; v < 4; ++v) {
      a[2 * v]     = lA[cur][wave * 16 + m][klo + 2 * v];
      a[2 * v + 1] = lA[cur][wave * 16 + m][klo + 2 * v + 1];
      a[8 + 2 * v]     = lA[cur][wave * 16 + m][16 + klo + 2 * v];
      a[8 + 2 * v + 1] = lA[cur][wave * 16 + m][16 + klo + 2 * v + 1];
    }
#pragma unroll
    for (int n = 0; n < 8; ++n) {
      v16h bf;
#pragma unroll
      for (int v = 0; v < 4; ++v) {
        bf[2 * v]     = lB[cur][n * 16 + m][klo + 2 * v];
        bf[2 * v + 1] = lB[cur][n * 16 + m][klo + 2 * v + 1];
        bf[8 + 2 * v]     = lB[cur][n * 16 + m][16 + klo + 2 * v];
        bf[8 + 2 * v + 1] = lB[cur][n * 16 + m][16 + klo + 2 * v + 1];
      }
      acc[n] = wmma_f16(a, bf, acc[n]);
    }
    __syncthreads();
  }

#pragma unroll
  for (int n = 0; n < 8; ++n)
#pragma unroll
    for (int r = 0; r < 8; ++r) {
      int row = bm + wave * 16 + ((lane < 16) ? r : r + 8);
      int col = bn + n * 16 + (lane & 15);
      C[(size_t)row * N + col] = acc[n][r];
    }
}

// ---------------- delta-rule chunked recurrence ----------------
// grid: B*H*NSL_ blocks, 256 threads (8 waves). Per-chunk math in fp32 WMMA.
__global__ __launch_bounds__(256) void delta_kernel(
    const float* __restrict__ qg, const float* __restrict__ kg,
    const float* __restrict__ vg, const float* __restrict__ betag,
    float* __restrict__ og)
{
  int blk = blockIdx.x;
  int slice = blk % NSL_;
  int h = (blk / NSL_) % H_;
  int b = blk / (NSL_ * H_);
  int dv0 = slice * DVS_;

  __shared__ float sS[DK_][DVS_ + 1];   // state slice
  __shared__ float sQ[CS_][DK_];        // q (raw, then scaled)
  __shared__ float sK[CS_][DK_];        // k (raw, then L2-normalized)
  __shared__ float sW[CS_][DK_];        // kb, then w = T@kb
  __shared__ float sV[CS_][DVS_ + 4];   // v (raw) -> v*beta -> u -> u_adj (pad keeps B128 align)
  __shared__ float sT[CS_][CS_ + 1];    // T, then attn
  __shared__ float sA[CS_][CS_ + 1];    // A = kb@k^T
  __shared__ float sBeta[CS_];
  __shared__ float sInv[CS_];

  int tid = threadIdx.x;
  int wave = tid >> 5, lane = tid & 31;
  int m = lane & 15;
  int khi = (lane < 16) ? 0 : 2;

  for (int i = tid; i < DK_ * DVS_; i += 256) sS[i / DVS_][i % DVS_] = 0.f;
  __syncthreads();

  const float qscale = rsqrtf((float)DK_);

  for (int c = 0; c < NC_; ++c) {
    int l0 = c * CS_;
    // ---- stage chunk (raw) via async global->LDS ----
    for (int i = tid; i < (CS_ * DK_) / 4; i += 256) {        // 512 B128 ops x2
      int r = i >> 5, c4 = (i & 31) * 4;
      size_t base = ((size_t)b * L_ + l0 + r) * KD_ + (size_t)h * DK_ + c4;
      cp16_lds(&sQ[r][c4], qg + base);
      cp16_lds(&sK[r][c4], kg + base);
    }
    for (int i = tid; i < (CS_ * DVS_) / 4; i += 256) {       // 128 B128 ops
      int r = i >> 3, c4 = (i & 7) * 4;
      cp16_lds(&sV[r][c4],
               vg + ((size_t)b * L_ + l0 + r) * VD_ + (size_t)h * DV_ + dv0 + c4);
    }
    if (tid < CS_) sBeta[tid] = betag[((size_t)b * L_ + l0 + tid) * H_ + h];
    WAIT_ASYNC(0);
    __syncthreads();

    // ---- k row norms (raw) ----
    if (tid < CS_) {
      float s = 1e-12f;
      for (int d = 0; d < DK_; ++d) { float kv = sK[tid][d]; s += kv * kv; }
      sInv[tid] = rsqrtf(s);
    }
    __syncthreads();
    // ---- transforms: q*=scale, k normalize, kb, v*=beta ----
    for (int i = tid; i < CS_ * DK_; i += 256) {
      int r = i >> 7, d = i & 127;
      sQ[r][d] *= qscale;
      float kn = sK[r][d] * sInv[r];
      sK[r][d] = kn;
      sW[r][d] = kn * sBeta[r];          // kb
    }
    for (int i = tid; i < CS_ * DVS_; i += 256) {
      int r = i / DVS_, n = i % DVS_;
      sV[r][n] *= sBeta[r];
    }
    __syncthreads();

    // ---- A = kb @ k^T (wave 0) ----
    if (wave == 0) {
      v8f acc = (v8f){0.f,0.f,0.f,0.f,0.f,0.f,0.f,0.f};
      for (int k0 = 0; k0 < DK_; k0 += 4) {
        v2f a, bf;
        a.x = sW[m][k0 + khi];     a.y = sW[m][k0 + khi + 1];
        bf.x = sK[m][k0 + khi];    bf.y = sK[m][k0 + khi + 1];  // B[kk][n] = k[n][kk]
        acc = wmma_f32(a, bf, acc);
      }
#pragma unroll
      for (int r = 0; r < 8; ++r) {
        int row = (lane < 16) ? r : r + 8;
        sA[row][m] = acc[r];
      }
    }
    __syncthreads();

    // ---- T = inv(I + strict_tril(A)), unit lower triangular ----
    if (wave == 0 && lane < CS_) {
      int j = lane;
      float t[CS_];
#pragma unroll
      for (int i = 0; i < CS_; ++i) t[i] = 0.f;
      t[j] = 1.f;
      for (int i = j + 1; i < CS_; ++i) {
        float s = 0.f;
        for (int mm = j; mm < i; ++mm) s += sA[i][mm] * t[mm];
        t[i] = -s;
      }
      for (int i = 0; i < CS_; ++i) sT[i][j] = t[i];
    }
    __syncthreads();

    // ---- w = T @ kb (8 waves, each its own 16-col slice, in-place in sW) ----
    {
      int d0 = wave * 16;
      v8f acc = (v8f){0.f,0.f,0.f,0.f,0.f,0.f,0.f,0.f};
      for (int k0 = 0; k0 < CS_; k0 += 4) {
        v2f a, bf;
        a.x = sT[m][k0 + khi];        a.y = sT[m][k0 + khi + 1];
        bf.x = sW[k0 + khi][d0 + m];  bf.y = sW[k0 + khi + 1][d0 + m];
        acc = wmma_f32(a, bf, acc);
      }
#pragma unroll
      for (int r = 0; r < 8; ++r) {
        int row = (lane < 16) ? r : r + 8;
        sW[row][d0 + m] = acc[r];
      }
    }
    // ---- u = T @ (v*beta) (waves 0..1, in-place in sV) ----
    if (wave < DVS_ / 16) {
      int n0 = wave * 16;
      v8f acc = (v8f){0.f,0.f,0.f,0.f,0.f,0.f,0.f,0.f};
      for (int k0 = 0; k0 < CS_; k0 += 4) {
        v2f a, bf;
        a.x = sT[m][k0 + khi];        a.y = sT[m][k0 + khi + 1];
        bf.x = sV[k0 + khi][n0 + m];  bf.y = sV[k0 + khi + 1][n0 + m];
        acc = wmma_f32(a, bf, acc);
      }
#pragma unroll
      for (int r = 0; r < 8; ++r) {
        int row = (lane < 16) ? r : r + 8;
        sV[row][n0 + m] = acc[r];
      }
    }
    __syncthreads();

    // ---- u_adj = u - w@S (waves 0..1) ; attn = causal(q@k^T) -> sT (wave 7) ----
    if (wave == 7) {
      v8f acc = (v8f){0.f,0.f,0.f,0.f,0.f,0.f,0.f,0.f};
      for (int k0 = 0; k0 < DK_; k0 += 4) {
        v2f a, bf;
        a.x = sQ[m][k0 + khi];   a.y = sQ[m][k0 + khi + 1];
        bf.x = sK[m][k0 + khi];  bf.y = sK[m][k0 + khi + 1];
        acc = wmma_f32(a, bf, acc);
      }
#pragma unroll
      for (int r = 0; r < 8; ++r) {
        int row = (lane < 16) ? r : r + 8;
        sT[row][m] = (m <= row) ? acc[r] : 0.f;
      }
    }
    if (wave < DVS_ / 16) {
      int n0 = wave * 16;
      v8f acc = (v8f){0.f,0.f,0.f,0.f,0.f,0.f,0.f,0.f};
      for (int k0 = 0; k0 < DK_; k0 += 4) {
        v2f a, bf;
        a.x = sW[m][k0 + khi];        a.y = sW[m][k0 + khi + 1];
        bf.x = sS[k0 + khi][n0 + m];  bf.y = sS[k0 + khi + 1][n0 + m];
        acc = wmma_f32(a, bf, acc);
      }
#pragma unroll
      for (int r = 0; r < 8; ++r) {
        int row = (lane < 16) ? r : r + 8;
        sV[row][n0 + m] = sV[row][n0 + m] - acc[r];   // own slice only
      }
    }
    __syncthreads();

    // ---- o = q@S + attn@u_adj (waves 0..1) and dS = k^T@u_adj (all waves) ----
    v8f oacc = (v8f){0.f,0.f,0.f,0.f,0.f,0.f,0.f,0.f};
    if (wave < DVS_ / 16) {
      int n0 = wave * 16;
      for (int k0 = 0; k0 < DK_; k0 += 4) {
        v2f a, bf;
        a.x = sQ[m][k0 + khi];        a.y = sQ[m][k0 + khi + 1];
        bf.x = sS[k0 + khi][n0 + m];  bf.y = sS[k0 + khi + 1][n0 + m];
        oacc = wmma_f32(a, bf, oacc);
      }
      for (int k0 = 0; k0 < CS_; k0 += 4) {
        v2f a, bf;
        a.x = sT[m][k0 + khi];        a.y = sT[m][k0 + khi + 1];
        bf.x = sV[k0 + khi][n0 + m];  bf.y = sV[k0 + khi + 1][n0 + m];
        oacc = wmma_f32(a, bf, oacc);
      }
    }
    v8f ds[DVS_ / 16];
    {
      int d0w = wave * 16;
#pragma unroll
      for (int t = 0; t < DVS_ / 16; ++t) {
        v8f acc = (v8f){0.f,0.f,0.f,0.f,0.f,0.f,0.f,0.f};
        for (int k0 = 0; k0 < CS_; k0 += 4) {
          v2f a, bf;
          a.x = sK[k0 + khi][d0w + m];        // A = k^T band
          a.y = sK[k0 + khi + 1][d0w + m];
          bf.x = sV[k0 + khi][t * 16 + m];
          bf.y = sV[k0 + khi + 1][t * 16 + m];
          acc = wmma_f32(a, bf, acc);
        }
        ds[t] = acc;
      }
    }
    __syncthreads();

    // ---- commit: S += dS ; store o ----
    {
      int d0w = wave * 16;
#pragma unroll
      for (int t = 0; t < DVS_ / 16; ++t)
#pragma unroll
        for (int r = 0; r < 8; ++r) {
          int row = (lane < 16) ? r : r + 8;
          sS[d0w + row][t * 16 + m] += ds[t][r];
        }
    }
    if (wave < DVS_ / 16) {
      int n0 = wave * 16;
#pragma unroll
      for (int r = 0; r < 8; ++r) {
        int row = (lane < 16) ? r : r + 8;
        og[(((size_t)b * L_ + l0 + row) * H_ + h) * DV_ + dv0 + n0 + m] = oacc[r];
      }
    }
    __syncthreads();
  }
}

// ---------------- RMS-norm + gate, emit f16 for final GEMM ----------------
__global__ __launch_bounds__(256) void norm_gate_kernel(
    const float* __restrict__ o, const float* __restrict__ g,
    const float* __restrict__ nw, _Float16* __restrict__ y)
{
  int row = blockIdx.x * 8 + (threadIdx.x >> 5);   // over B*L*H
  int lane = threadIdx.x & 31;
  if (row >= B_ * L_ * H_) return;
  const float* orow = o + (size_t)row * DV_;
  const float* grow = g + (size_t)row * DV_;       // g is [B,L,H*DV], same row stride
  float ss = 0.f;
  for (int i = lane; i < DV_; i += 32) { float v = orow[i]; ss += v * v; }
  for (int off = 16; off; off >>= 1) ss += __shfl_down(ss, off, 32);
  ss = __shfl(ss, 0, 32);
  float inv = rsqrtf(ss / (float)DV_ + 1e-5f);
  for (int i = lane; i < DV_; i += 32)
    y[(size_t)row * DV_ + i] = (_Float16)(orow[i] * inv * nw[i] * silu_f(grow[i]));
}

// ---------------- host launcher ----------------
extern "C" void kernel_launch(void* const* d_in, const int* in_sizes, int n_in,
                              void* d_out, int out_size, void* d_ws, size_t ws_size,
                              hipStream_t stream)
{
  const float* hs     = (const float*)d_in[0];
  const float* conv_w = (const float*)d_in[1];
  const float* Wq     = (const float*)d_in[2];
  const float* Wk     = (const float*)d_in[3];
  const float* Wv     = (const float*)d_in[4];
  const float* Wb     = (const float*)d_in[5];
  const float* Wg     = (const float*)d_in[6];
  const float* Wo     = (const float*)d_in[7];
  const float* norm_w = (const float*)d_in[8];
  float* out = (float*)d_out;

  char* ws = (char*)d_ws;
  size_t off = 0;
  auto alloc = [&](size_t bytes) -> char* {
    char* p = ws + off;
    off = (off + bytes + 255) & ~(size_t)255;
    return p;
  };
  const size_t ROWS = (size_t)B_ * L_;            // 8192
  float*     x_f  = (float*)    alloc(ROWS * D_ * 4);
  _Float16*  x_h  = (_Float16*) alloc(ROWS * D_ * 2);
  _Float16*  wq_h = (_Float16*) alloc((size_t)KD_ * D_ * 2);
  _Float16*  wk_h = (_Float16*) alloc((size_t)KD_ * D_ * 2);
  _Float16*  wv_h = (_Float16*) alloc((size_t)VD_ * D_ * 2);
  _Float16*  wg_h = (_Float16*) alloc((size_t)VD_ * D_ * 2);
  _Float16*  wo_h = (_Float16*) alloc((size_t)D_ * VD_ * 2);
  float*     qb   = (float*)    alloc(ROWS * KD_ * 4);
  float*     kb   = (float*)    alloc(ROWS * KD_ * 4);
  float*     vb   = (float*)    alloc(ROWS * VD_ * 4);
  float*     gb   = (float*)    alloc(ROWS * VD_ * 4);
  float*     beta = (float*)    alloc(ROWS * H_ * 4);
  float*     ob   = (float*)    alloc(ROWS * VD_ * 4);
  _Float16*  y_h  = (_Float16*) alloc(ROWS * VD_ * 2);
  (void)ws_size; (void)n_in; (void)in_sizes; (void)out_size;

  // 1) conv + SiLU
  conv_silu_kernel<<<(B_ * L_ * D_) / 256, 256, 0, stream>>>(hs, conv_w, x_f, x_h);

  // 2) cast weights to f16
  cvt_f32_f16_kernel<<<(KD_ * D_) / 256, 256, 0, stream>>>(Wq, wq_h, KD_ * D_);
  cvt_f32_f16_kernel<<<(KD_ * D_) / 256, 256, 0, stream>>>(Wk, wk_h, KD_ * D_);
  cvt_f32_f16_kernel<<<(VD_ * D_) / 256, 256, 0, stream>>>(Wv, wv_h, VD_ * D_);
  cvt_f32_f16_kernel<<<(VD_ * D_) / 256, 256, 0, stream>>>(Wg, wg_h, VD_ * D_);
  cvt_f32_f16_kernel<<<(D_ * VD_) / 256, 256, 0, stream>>>(Wo, wo_h, D_ * VD_);

  // 3) projections (WMMA f16 GEMMs) + beta
  dim3 blk(256);
  gemm_f16_kernel<<<dim3(KD_ / 128, ROWS / 128), blk, 0, stream>>>(x_h, wq_h, qb, ROWS, KD_, D_);
  gemm_f16_kernel<<<dim3(KD_ / 128, ROWS / 128), blk, 0, stream>>>(x_h, wk_h, kb, ROWS, KD_, D_);
  gemm_f16_kernel<<<dim3(VD_ / 128, ROWS / 128), blk, 0, stream>>>(x_h, wv_h, vb, ROWS, VD_, D_);
  gemm_f16_kernel<<<dim3(VD_ / 128, ROWS / 128), blk, 0, stream>>>(x_h, wg_h, gb, ROWS, VD_, D_);
  beta_kernel<<<(ROWS + 7) / 8, 256, 0, stream>>>(x_f, Wb, beta);

  // 4) delta-rule recurrence (fp32 WMMA)
  delta_kernel<<<B_ * H_ * NSL_, 256, 0, stream>>>(qb, kb, vb, beta, ob);

  // 5) norm + gate, then output GEMM
  norm_gate_kernel<<<(B_ * L_ * H_ + 7) / 8, 256, 0, stream>>>(ob, gb, norm_w, y_h);
  gemm_f16_kernel<<<dim3(D_ / 128, ROWS / 128), blk, 0, stream>>>(y_h, wo_h, out, ROWS, D_, VD_);
}
